// CGEncoder_18983755448830
// MI455X (gfx1250) — matrix operand order repaired and checked by reading, not compiled, not run on hardware
//
#include <hip/hip_runtime.h>
#include <math.h>

#define NA 20000
#define NCG 1000
#define NE 200000
#define FEATC 128
#define NRBF 20
#define NCONV 3
#define CUTOFF_F 5.0f
#define EPSC 1e-15f
#define PI_F 3.14159265358979323846f

typedef _Float16 h16;
typedef __attribute__((ext_vector_type(16))) _Float16 v16h;
typedef __attribute__((ext_vector_type(8)))  _Float16 v8h;
typedef __attribute__((ext_vector_type(8)))  float    v8f;

__device__ __forceinline__ float swish_f(float x) { return x / (1.0f + __expf(-x)); }

// ---------------------------------------------------------------------------
// WMMA GEMM: C[M,N] = act(A[M,K] @ W[K,N] + bias), f16 inputs / f32 accum.
// Block = 256 threads (8 waves). Tile: 128(M) x 64(N), K step 32.
// LDS tiles are stored in *fragment order* so each lane's v16h is a
// contiguous 32-byte run -> ds_load_b128 pairs instead of ds_load_u16 storms.
//   As[(wave*32 + lane)*16 + i] = A[m0 + wave*16 + (lane&15)]
//                                  [k0 + 16*(i>>3) + 8*(lane>>4) + (i&7)]
//   Bs[(nt*32 + lane)*16 + i]   = W[k0 + 16*(i>>3) + 8*(lane>>4) + (i&7)]
//                                  [n0 + nt*16 + (lane&15)]
// ---------------------------------------------------------------------------
__global__ __launch_bounds__(256) void gemm_bias_act(
    const float* __restrict__ A, const float* __restrict__ W,
    const float* __restrict__ bias, float* __restrict__ C,
    int M, int N, int K, int ldw, int ldc, int act)
{
  __shared__ alignas(32) h16 As[128 * 32]; // [8 waves][32 lanes][16 halves]
  __shared__ alignas(32) h16 Bs[32 * 64];  // [4 nt]   [32 lanes][16 halves]
  const int tid  = threadIdx.x;
  const int wave = tid >> 5;
  const int lane = tid & 31;
  const int hi   = (lane >> 4) & 1;
  const int l15  = lane & 15;
  const int m0   = blockIdx.x * 128;
  const int n0   = blockIdx.y * 64;

  v8f acc[4] = {};

  for (int k0 = 0; k0 < K; k0 += 32) {
    // ---- Stage A: 512 chunks of 8 halves, contiguous in both global & LDS.
    // chunk c: i8=c&1, ln=(c>>1)&31, wv=c>>6 ; m=wv*16+(ln&15),
    //          kk = 16*i8 + 8*(ln>>4) + j  (j=0..7 contiguous in global)
#pragma unroll
    for (int t = 0; t < 2; ++t) {
      int c   = tid + t * 256;
      int i8  = c & 1;
      int ln  = (c >> 1) & 31;
      int wv  = c >> 6;
      int m   = wv * 16 + (ln & 15);
      int kkb = i8 * 16 + ((ln >> 4) << 3);
      int gr  = m0 + m;
      v8h pack;
      if (gr < M) {
        const float4* ap =
            reinterpret_cast<const float4*>(A + (size_t)gr * K + k0 + kkb);
        float4 x = ap[0];
        float4 y = ap[1];
        pack[0] = (h16)x.x; pack[1] = (h16)x.y; pack[2] = (h16)x.z; pack[3] = (h16)x.w;
        pack[4] = (h16)y.x; pack[5] = (h16)y.y; pack[6] = (h16)y.z; pack[7] = (h16)y.w;
      } else {
        pack = (v8h){};
      }
      *reinterpret_cast<v8h*>(&As[c * 8]) = pack;
    }
    // ---- Stage B: 256 chunks of 8 halves (global column-strided; W tile is
    // small and block-shared -> cache resident). One ds_store_b128 per chunk.
    {
      int c   = tid;
      int i8  = c & 1;
      int ln  = (c >> 1) & 31;
      int nt  = c >> 6;
      int col = n0 + nt * 16 + (ln & 15);
      int kb  = k0 + i8 * 16 + ((ln >> 4) << 3);
      v8h pack;
#pragma unroll
      for (int j = 0; j < 8; ++j)
        pack[j] = (h16)W[(size_t)(kb + j) * ldw + col];
      *reinterpret_cast<v8h*>(&Bs[c * 8]) = pack;
    }
    __syncthreads();

    const v16h a = *reinterpret_cast<const v16h*>(&As[(wave * 32 + lane) * 16]);
#pragma unroll
    for (int nt = 0; nt < 4; ++nt) {
      const v16h b = *reinterpret_cast<const v16h*>(&Bs[(nt * 32 + lane) * 16]);
      acc[nt] = __builtin_amdgcn_wmma_f32_16x16x32_f16(
          false, a, false, b, (short)0, acc[nt], false, false);
    }
    __syncthreads();
  }

  // Epilogue: C layout VGPR r -> row r + 8*hi, lane -> col
#pragma unroll
  for (int nt = 0; nt < 4; ++nt) {
    const int coln = n0 + nt * 16 + l15;
    const float bv = bias ? bias[coln] : 0.f;
#pragma unroll
    for (int r = 0; r < 8; ++r) {
      int grow = m0 + wave * 16 + r + (hi ? 8 : 0);
      if (grow < M) {
        float val = acc[nt][r] + bv;
        if (act) val = swish_f(val);
        C[(size_t)grow * ldc + coln] = val;
      }
    }
  }
}

// ---------------------------------------------------------------------------
// Elementwise / setup kernels
// ---------------------------------------------------------------------------
__global__ void zero_f32(float* __restrict__ p, long long n) {
  long long i = (long long)blockIdx.x * blockDim.x + threadIdx.x;
  if (i < n) p[i] = 0.f;
}
__global__ void add_inplace(float* __restrict__ d, const float* __restrict__ s, long long n) {
  long long i = (long long)blockIdx.x * blockDim.x + threadIdx.x;
  if (i < n) d[i] += s[i];
}
__global__ void gather_embed(const int* __restrict__ z, const float* __restrict__ embed,
                             float* __restrict__ s) {
  long long i = (long long)blockIdx.x * blockDim.x + threadIdx.x;
  if (i >= (long long)NA * FEATC) return;
  int n = (int)(i >> 7), f = (int)(i & 127);
  s[i] = embed[(size_t)z[n] * FEATC + f];
}
__global__ void count_atoms(const int* __restrict__ mapping, float* __restrict__ cnt) {
  int n = blockIdx.x * blockDim.x + threadIdx.x;
  if (n < NA) atomicAdd(&cnt[mapping[n]], 1.0f);
}
__global__ void make_invcnt(const float* __restrict__ cnt, float* __restrict__ inv) {
  int i = blockIdx.x * blockDim.x + threadIdx.x;
  if (i < NCG) inv[i] = 1.0f / fmaxf(cnt[i], 1.0f);
}
__global__ void edge_geom(const float* __restrict__ xyz, const int* __restrict__ nbr,
                          float* __restrict__ dij, float* __restrict__ uij) {
  int e = blockIdx.x * blockDim.x + threadIdx.x;
  if (e >= NE) return;
  int s = nbr[2 * e], d = nbr[2 * e + 1];
  float r0 = xyz[3 * d]     - xyz[3 * s];
  float r1 = xyz[3 * d + 1] - xyz[3 * s + 1];
  float r2 = xyz[3 * d + 2] - xyz[3 * s + 2];
  float dist = sqrtf(r0 * r0 + r1 * r1 + r2 * r2 + 3.0f * EPSC);
  dij[e] = dist;
  uij[3 * e] = r0 / dist; uij[3 * e + 1] = r1 / dist; uij[3 * e + 2] = r2 / dist;
}
__global__ void atom_geom(const float* __restrict__ xyz, const float* __restrict__ cg_xyz,
                          const int* __restrict__ mapping,
                          float* __restrict__ diI, float* __restrict__ uiI) {
  int n = blockIdx.x * blockDim.x + threadIdx.x;
  if (n >= NA) return;
  int I = mapping[n];
  float r0 = xyz[3 * n]     - cg_xyz[3 * I];
  float r1 = xyz[3 * n + 1] - cg_xyz[3 * I + 1];
  float r2 = xyz[3 * n + 2] - cg_xyz[3 * I + 2];
  float dist = sqrtf(r0 * r0 + r1 * r1 + r2 * r2 + 3.0f * EPSC);
  diI[n] = dist;
  uiI[3 * n] = r0 / dist; uiI[3 * n + 1] = r1 / dist; uiI[3 * n + 2] = r2 / dist;
}
__global__ void vnorm_kernel(const float* __restrict__ vv, float* __restrict__ vn) {
  long long i = (long long)blockIdx.x * blockDim.x + threadIdx.x;
  if (i >= (long long)NA * FEATC) return;
  long long n = i >> 7; int f = (int)(i & 127);
  size_t b = (size_t)n * 384 + f;
  float s = 0.f;
#pragma unroll
  for (int d = 0; d < 3; ++d) { float x = vv[b + d * 128]; s += x * x + EPSC; }
  vn[i] = sqrtf(s);
}
__global__ void concat_kernel(const float* __restrict__ s, const float* __restrict__ vn,
                              float* __restrict__ cat) {
  long long i = (long long)blockIdx.x * blockDim.x + threadIdx.x;
  if (i >= (long long)NA * 256) return;
  long long n = i >> 8; int j = (int)(i & 255);
  cat[i] = (j < 128) ? s[(size_t)n * 128 + j] : vn[(size_t)n * 128 + (j - 128)];
}
__global__ void update_sv(const float* __restrict__ stack, const float* __restrict__ uv,
                          const float* __restrict__ vv, float* __restrict__ v3,
                          float* __restrict__ s) {
  long long i = (long long)blockIdx.x * blockDim.x + threadIdx.x;
  if (i >= (long long)NA * FEATC) return;
  long long n = i >> 7; int f = (int)(i & 127);
  const float* st = stack + (size_t)n * 384;
  float avv = st[f], asv = st[128 + f], ass = st[256 + f];
  size_t b = (size_t)n * 384 + f;
  float dot = 0.f;
#pragma unroll
  for (int d = 0; d < 3; ++d) {
    float u = uv[b + d * 128], w = vv[b + d * 128];
    dot += u * w;
    v3[b + d * 128] += u * avv;
  }
  s[i] += dot * asv + ass;
}
__global__ void scatter_s(const float* __restrict__ s, const int* __restrict__ mapping,
                          const float* __restrict__ invcnt, float* __restrict__ SI) {
  long long i = (long long)blockIdx.x * blockDim.x + threadIdx.x;
  if (i >= (long long)NA * FEATC) return;
  int n = (int)(i >> 7), f = (int)(i & 127);
  int I = mapping[n];
  atomicAdd(&SI[(size_t)I * FEATC + f], s[i] * invcnt[I]);
}

// ---------------------------------------------------------------------------
// Fused edge message: RBF embed (LDS-cached distW) + gather phi[dst], v[dst],
// scatter-add into dsacc[src], dvacc[src] via f32 global atomics.
// ---------------------------------------------------------------------------
__global__ __launch_bounds__(128) void edge_message(
    const int* __restrict__ nbr, const float* __restrict__ dij,
    const float* __restrict__ uij, const float* __restrict__ phi,
    const float* __restrict__ v3, const float* __restrict__ distW,
    const float* __restrict__ distb,
    float* __restrict__ dsacc, float* __restrict__ dvacc)
{
  __shared__ float sW[NRBF * 384];
  __shared__ float sB[384];
  __shared__ float sR[NRBF];
  const int tid = threadIdx.x;
  for (int i = tid; i < NRBF * 384; i += 128) sW[i] = distW[i];
  for (int i = tid; i < 384; i += 128) sB[i] = distb[i];
  __syncthreads();

  const int EPB = 16;
  for (int it = 0; it < EPB; ++it) {
    int e = blockIdx.x * EPB + it;
    if (e >= NE) break;
    float d = dij[e];
    if (tid < NRBF) sR[tid] = __sinf((float)(tid + 1) * PI_F * d / CUTOFF_F) / d;
    __syncthreads();
    float env = 0.5f * (__cosf(PI_F * d / CUTOFF_F) + 1.0f) * (d < CUTOFF_F ? 1.f : 0.f);
    int src = nbr[2 * e], dst = nbr[2 * e + 1];
    float u0 = uij[3 * e], u1 = uij[3 * e + 1], u2 = uij[3 * e + 2];
    const float* ph = phi + (size_t)dst * 384;
    const float* vd = v3  + (size_t)dst * 384;
    float w[3];
#pragma unroll
    for (int g = 0; g < 3; ++g) {
      int j = g * 128 + tid;
      float acc = sB[j];
#pragma unroll
      for (int k = 0; k < NRBF; ++k) acc += sR[k] * sW[k * 384 + j];
      w[g] = acc * env;
    }
    float inv0 = ph[tid] * w[0];
    float inv1 = ph[128 + tid] * w[1];
    float inv2 = ph[256 + tid] * w[2];
    atomicAdd(&dsacc[(size_t)src * 128 + tid], inv1);
    float* dva = dvacc + (size_t)src * 384;
    atomicAdd(&dva[tid],       inv2 * u0 + inv0 * vd[tid]);
    atomicAdd(&dva[128 + tid], inv2 * u1 + inv0 * vd[128 + tid]);
    atomicAdd(&dva[256 + tid], inv2 * u2 + inv0 * vd[256 + tid]);
    __syncthreads();
  }
}

// ---------------------------------------------------------------------------
// CG message (only the scalar channel survives to the output): per atom,
// w_cg middle slice * phi_cg middle slice -> scaled atomic scatter into S_I.
// ---------------------------------------------------------------------------
__global__ __launch_bounds__(128) void cg_message(
    const float* __restrict__ phimid, const float* __restrict__ diI,
    const int* __restrict__ mapping, const float* __restrict__ invcnt,
    const float* __restrict__ distW, const float* __restrict__ distb,
    float* __restrict__ SI)
{
  __shared__ float sW[NRBF * 128];
  __shared__ float sB[128];
  __shared__ float sR[NRBF];
  const int tid = threadIdx.x;
  for (int i = tid; i < NRBF * 128; i += 128) {
    int k = i >> 7, f = i & 127;
    sW[i] = distW[k * 384 + 128 + f];
  }
  sB[tid] = distb[128 + tid];
  __syncthreads();

  const int APB = 16;
  for (int it = 0; it < APB; ++it) {
    int n = blockIdx.x * APB + it;
    if (n >= NA) break;
    float d = diI[n];
    if (tid < NRBF) sR[tid] = __sinf((float)(tid + 1) * PI_F * d / CUTOFF_F) / d;
    __syncthreads();
    float env = 0.5f * (__cosf(PI_F * d / CUTOFF_F) + 1.0f) * (d < CUTOFF_F ? 1.f : 0.f);
    float acc = sB[tid];
#pragma unroll
    for (int k = 0; k < NRBF; ++k) acc += sR[k] * sW[k * 128 + tid];
    float w = acc * env;
    float inv1 = phimid[(size_t)n * 128 + tid] * w;
    int I = mapping[n];
    atomicAdd(&SI[(size_t)I * 128 + tid], inv1 * invcnt[I]);
    __syncthreads();
  }
}

// ---------------------------------------------------------------------------
// Host driver
// ---------------------------------------------------------------------------
static inline void launch_gemm(const float* A, const float* W, const float* b, float* C,
                               int M, int N, int K, int ldw, int ldc, int act,
                               hipStream_t stream) {
  dim3 grid((M + 127) / 128, N / 64);
  gemm_bias_act<<<grid, 256, 0, stream>>>(A, W, b, C, M, N, K, ldw, ldc, act);
}
static inline void launch_zero(float* p, long long n, hipStream_t stream) {
  zero_f32<<<(unsigned)((n + 255) / 256), 256, 0, stream>>>(p, n);
}
static inline void launch_add(float* d, const float* s, long long n, hipStream_t stream) {
  add_inplace<<<(unsigned)((n + 255) / 256), 256, 0, stream>>>(d, s, n);
}

extern "C" void kernel_launch(void* const* d_in, const int* in_sizes, int n_in,
                              void* d_out, int out_size, void* d_ws, size_t ws_size,
                              hipStream_t stream) {
  const int*   z        = (const int*)d_in[0];
  const float* xyz      = (const float*)d_in[1];
  const float* cg_xyz   = (const float*)d_in[2];
  const int*   mapping  = (const int*)d_in[3];
  const int*   nbr      = (const int*)d_in[4];
  const float* embed    = (const float*)d_in[5];
  const float* msg_W1   = (const float*)d_in[6];
  const float* msg_b1   = (const float*)d_in[7];
  const float* msg_W2   = (const float*)d_in[8];
  const float* msg_b2   = (const float*)d_in[9];
  const float* msg_dW   = (const float*)d_in[10];
  const float* msg_db   = (const float*)d_in[11];
  const float* upd_U    = (const float*)d_in[12];
  const float* upd_V    = (const float*)d_in[13];
  const float* upd_W1   = (const float*)d_in[14];
  const float* upd_b1   = (const float*)d_in[15];
  const float* upd_W2   = (const float*)d_in[16];
  const float* upd_b2   = (const float*)d_in[17];
  const float* cg_W1    = (const float*)d_in[18];
  const float* cg_b1    = (const float*)d_in[19];
  const float* cg_W2    = (const float*)d_in[20];
  const float* cg_b2    = (const float*)d_in[21];
  const float* cg_dW    = (const float*)d_in[22];
  const float* cg_db    = (const float*)d_in[23];
  float* SI = (float*)d_out;

  // Workspace layout (floats)
  float* ws = (float*)d_ws;
  size_t o = 0;
  float* s      = ws + o; o += (size_t)NA * 128;
  float* v3     = ws + o; o += (size_t)NA * 384;   // [N,3,128]
  float* dsacc  = ws + o; o += (size_t)NA * 128;
  float* dvacc  = ws + o; o += (size_t)NA * 384;
  float* bufA   = ws + o; o += (size_t)NA * 128;   // hidden layer
  float* bufP   = ws + o; o += (size_t)NA * 384;   // phi / stack / phi_cg_mid
  float* bufC   = ws + o; o += (size_t)NA * 256;   // concat(s, vnorm)
  float* uvbuf  = ws + o; o += (size_t)NA * 384;
  float* vvbuf  = ws + o; o += (size_t)NA * 384;
  float* vnbuf  = ws + o; o += (size_t)NA * 128;
  float* dij    = ws + o; o += (size_t)NE;
  float* uij    = ws + o; o += (size_t)NE * 3;
  float* diI    = ws + o; o += (size_t)NA;
  float* uiI    = ws + o; o += (size_t)NA * 3;
  float* cnt    = ws + o; o += (size_t)NCG;
  float* invcnt = ws + o; o += (size_t)NCG;
  (void)ws_size; (void)in_sizes; (void)n_in; (void)out_size;

  // ---- Setup ----
  launch_zero(SI, (long long)NCG * 128, stream);
  launch_zero(v3, (long long)NA * 384, stream);
  launch_zero(cnt, NCG, stream);
  count_atoms<<<(NA + 255) / 256, 256, 0, stream>>>(mapping, cnt);
  make_invcnt<<<(NCG + 255) / 256, 256, 0, stream>>>(cnt, invcnt);
  gather_embed<<<(unsigned)(((long long)NA * 128 + 255) / 256), 256, 0, stream>>>(z, embed, s);
  edge_geom<<<(NE + 255) / 256, 256, 0, stream>>>(xyz, nbr, dij, uij);
  atom_geom<<<(NA + 255) / 256, 256, 0, stream>>>(xyz, cg_xyz, mapping, diI, uiI);

  for (int i = 0; i < NCONV; ++i) {
    const float* mW1 = msg_W1 + (size_t)i * 128 * 128;
    const float* mb1 = msg_b1 + (size_t)i * 128;
    const float* mW2 = msg_W2 + (size_t)i * 128 * 384;
    const float* mb2 = msg_b2 + (size_t)i * 384;
    const float* mdW = msg_dW + (size_t)i * NRBF * 384;
    const float* mdb = msg_db + (size_t)i * 384;
    const float* uU  = upd_U  + (size_t)i * 128 * 128;
    const float* uV  = upd_V  + (size_t)i * 128 * 128;
    const float* uW1 = upd_W1 + (size_t)i * 256 * 128;
    const float* ub1 = upd_b1 + (size_t)i * 128;
    const float* uW2 = upd_W2 + (size_t)i * 128 * 384;
    const float* ub2 = upd_b2 + (size_t)i * 384;
    const float* cW1 = cg_W1  + (size_t)i * 128 * 128;
    const float* cb1 = cg_b1  + (size_t)i * 128;
    const float* cW2 = cg_W2  + (size_t)i * 128 * 384;
    const float* cb2 = cg_b2  + (size_t)i * 384;
    const float* cdW = cg_dW  + (size_t)i * NRBF * 384;
    const float* cdb = cg_db  + (size_t)i * 384;

    // phi = swish(s@W1+b1)@W2+b2
    launch_gemm(s,    mW1, mb1, bufA, NA, 128, 128, 128, 128, 1, stream);
    launch_gemm(bufA, mW2, mb2, bufP, NA, 384, 128, 384, 384, 0, stream);

    // edge message -> dsacc, dvacc (double-buffered: reads old v3)
    launch_zero(dsacc, (long long)NA * 128, stream);
    launch_zero(dvacc, (long long)NA * 384, stream);
    edge_message<<<(NE + 15) / 16, 128, 0, stream>>>(nbr, dij, uij, bufP, v3,
                                                     mdW, mdb, dsacc, dvacc);
    launch_add(s,  dsacc, (long long)NA * 128, stream);
    launch_add(v3, dvacc, (long long)NA * 384, stream);

    // u_v / v_v : [3N,128] x [128,128]
    launch_gemm(v3, uU, nullptr, uvbuf, NA * 3, 128, 128, 128, 128, 0, stream);
    launch_gemm(v3, uV, nullptr, vvbuf, NA * 3, 128, 128, 128, 128, 0, stream);

    vnorm_kernel<<<(unsigned)(((long long)NA * 128 + 255) / 256), 256, 0, stream>>>(vvbuf, vnbuf);
    concat_kernel<<<(unsigned)(((long long)NA * 256 + 255) / 256), 256, 0, stream>>>(s, vnbuf, bufC);

    // stack = swish(cat@W1+b1)@W2+b2
    launch_gemm(bufC, uW1, ub1, bufA, NA, 128, 256, 128, 128, 1, stream);
    launch_gemm(bufA, uW2, ub2, bufP, NA, 384, 128, 384, 384, 0, stream);

    update_sv<<<(unsigned)(((long long)NA * 128 + 255) / 256), 256, 0, stream>>>(
        bufP, uvbuf, vvbuf, v3, s);

    if (i == 0) {
      scatter_s<<<(unsigned)(((long long)NA * 128 + 255) / 256), 256, 0, stream>>>(
          s, mapping, invcnt, SI);
    }

    // CG path: only middle 128 cols of phi_cg survive to S_I
    launch_gemm(s,    cW1,       cb1,       bufA, NA, 128, 128, 128, 128, 1, stream);
    launch_gemm(bufA, cW2 + 128, cb2 + 128, bufP, NA, 128, 128, 384, 128, 0, stream);
    cg_message<<<(NA + 15) / 16, 128, 0, stream>>>(bufP, diI, mapping, invcnt,
                                                   cdW, cdb, SI);
  }
}
